// GAT_MultiLayer_51153060495544
// MI455X (gfx1250) — matrix operand order repaired and checked by reading, not compiled, not run on hardware
//
#include <hip/hip_runtime.h>

// ---------------------------------------------------------------------------
// GAT 3-layer forward for MI455X (gfx1250, wave32, WMMA).
// GEMMs: v_wmma_f32_16x16x32_f16, A from global (b128 loads, no predication),
// B staged fp32->f16 into LDS in fragment order (ds_load_b128 per wave).
// Edge phase: L2-resident fp32 gather + global_atomic_add_f32 scatter.
// ---------------------------------------------------------------------------

typedef __attribute__((ext_vector_type(16))) _Float16 v16h;
typedef __attribute__((ext_vector_type(8)))  float    v8f;

#define N_NODES 50000
#define N_EDGES 800000
#define NEG_SLOPE 0.2f
#define EPS_F 1e-16f

static inline long cdivl(long a, long b) { return (a + b - 1) / b; }

// Order-preserving float->uint mapping so unsigned atomicMax == float max.
__device__ __forceinline__ unsigned flip_f32(float f) {
  unsigned u = __float_as_uint(f);
  return (u & 0x80000000u) ? ~u : (u | 0x80000000u);
}
__device__ __forceinline__ float unflip_f32(unsigned u) {
  unsigned v = (u & 0x80000000u) ? (u ^ 0x80000000u) : ~u;
  return __uint_as_float(v);
}

// --------------------------- generic fill ----------------------------------
__global__ void fill_u32(unsigned* __restrict__ p, unsigned v, long n) {
  long i = (long)blockIdx.x * blockDim.x + threadIdx.x;
  if (i < n) p[i] = v;
}

// ---- zero-pad a [K,NC] weight matrix to [K,NCpad] (for the 2-col layer) ----
__global__ void pad_w(const float* __restrict__ W, float* __restrict__ Wp,
                      int K, int NC, int NCpad) {
  int i = blockIdx.x * blockDim.x + threadIdx.x;  // over K*NCpad
  if (i >= K * NCpad) return;
  int k = i / NCpad, c = i % NCpad;
  Wp[i] = (c < NC) ? W[k * NC + c] : 0.f;
}

// --------------------------- WMMA GEMM -------------------------------------
// C[M,NCout] = A[M,K] @ B[K,NCpad] (cols >= NCout are zero padding).
// Requirements: M % 16 == 0, K % 32 == 0, NCpad % 16 == 0, K <= 256.
// Grid: (ceil(M/16 / wavesPerBlock), NCpad/16); 8 wave32 per block, each wave
// owns one 16x16 output tile; all waves in a block share one B column panel.
//
// Fragment layouts (ISA 7.12.2, wave32):
//  A 16x32 f16 : lane L holds row M=L&15; elems e<8 -> K=(L>=16?8:0)+e,
//                e>=8 -> K=16+(L>=16?8:0)+(e-8)  => two 8-contiguous-K runs.
//  B 32x16 f16 : lane L holds col N=L&15; elem e -> K=(L>=16?8:0)+(e&7)+(e>=8?16:0) (same as A).
//  C/D 16x16 f32: lane L col N=L&15; acc[v] -> row M = v + 8*(L>=16).
__global__ void gemm_wmma_f16(const float* __restrict__ A,
                              const float* __restrict__ Bm,
                              float* __restrict__ C,
                              int M, int K, int NCpad, int NCout) {
  __shared__ __align__(32) _Float16 ldsB[4096];  // (K/32)*512, K<=256 -> 8KB
  const int tid  = threadIdx.x;
  const int lane = tid & 31;
  const int wave = tid >> 5;
  const int nt   = blockIdx.y;
  const int mt   = blockIdx.x * (blockDim.x >> 5) + wave;
  const int nTm  = M >> 4;
  const int l15  = lane & 15;
  const int hi   = lane >> 4;  // 0 or 1
  const int nchunks = K >> 5;

  // Stage B panel (cols nt*16..+15) into LDS, pre-swizzled to fragment order:
  // lds[c*512 + L*16 + e] = B[(c*32 + (L>=16)*8 + (e&7) + (e>=8)*16) * NCpad + nt*16 + (L&15)]
  for (int idx = tid; idx < (nchunks << 9); idx += blockDim.x) {
    int c = idx >> 9;
    int r = idx & 511;
    int L = r >> 4;
    int e = r & 15;
    int k = (c << 5) + ((L >> 4) << 3) + (e & 7) + ((e >> 3) << 4);
    int col = nt * 16 + (L & 15);
    ldsB[idx] = (_Float16)Bm[(size_t)k * NCpad + col];
  }
  __syncthreads();

  if (mt < nTm) {  // wave-uniform; EXEC all-1s inside (WMMA requirement)
    const float* arowp = A + (size_t)(mt * 16 + l15) * K;
    const int ka = hi * 8;
    v8f acc = {};
    for (int c = 0; c < nchunks; ++c) {
      const int k0 = c << 5;
      // A fragment: 4 x b128, fully unpredicated (M,K multiples of tile)
      const float4* ap0 = (const float4*)(arowp + k0 + ka);
      const float4* ap1 = (const float4*)(arowp + k0 + 16 + ka);
      float4 a0 = ap0[0], a1 = ap0[1];
      float4 a2 = ap1[0], a3 = ap1[1];
      v16h a;
      a[0]  = (_Float16)a0.x; a[1]  = (_Float16)a0.y;
      a[2]  = (_Float16)a0.z; a[3]  = (_Float16)a0.w;
      a[4]  = (_Float16)a1.x; a[5]  = (_Float16)a1.y;
      a[6]  = (_Float16)a1.z; a[7]  = (_Float16)a1.w;
      a[8]  = (_Float16)a2.x; a[9]  = (_Float16)a2.y;
      a[10] = (_Float16)a2.z; a[11] = (_Float16)a2.w;
      a[12] = (_Float16)a3.x; a[13] = (_Float16)a3.y;
      a[14] = (_Float16)a3.z; a[15] = (_Float16)a3.w;
      // B fragment: one contiguous 32B LDS read per wave (2 x ds_load_b128)
      v16h b = *(const v16h*)(ldsB + (c << 9) + lane * 16);
      acc = __builtin_amdgcn_wmma_f32_16x16x32_f16(
          /*neg_a=*/false, a, /*neg_b=*/false, b,
          /*c_mod=*/(short)0, acc, /*reuse_a=*/false, /*reuse_b=*/false);
    }
    const int coln = nt * 16 + l15;
    if (coln < NCout) {
#pragma unroll
      for (int v = 0; v < 8; ++v) {
        int m = mt * 16 + hi * 8 + v;
        C[(size_t)m * NCout + coln] = acc[v];
      }
    }
  }
}

// ----------------- per-node, per-head attention coefficients ---------------
__global__ void attn_coef(const float* __restrict__ h,
                          const float* __restrict__ a_src,
                          const float* __restrict__ a_dst,
                          float* __restrict__ as, float* __restrict__ ad,
                          int n, int H, int C) {
  int i = blockIdx.x * blockDim.x + threadIdx.x;  // over n*H
  if (i >= n * H) return;
  int hd = i % H;
  const float* hp = h + (size_t)i * C;            // i*C == (node*H+hd)*C
  float s = 0.f, d = 0.f;
  for (int c = 0; c < C; ++c) {
    float v = hp[c];
    s += v * a_src[hd * C + c];
    d += v * a_dst[hd * C + c];
  }
  as[i] = s;
  ad[i] = d;
}

// -------- edge logits: e = leakyrelu(as[src]+ad[dst]); segment max ---------
__global__ void edge_logits(const int* __restrict__ src,
                            const int* __restrict__ dst,
                            const float* __restrict__ as,
                            const float* __restrict__ ad,
                            float* __restrict__ ebuf,
                            unsigned* __restrict__ mflip,
                            int E, int H) {
  long i = (long)blockIdx.x * blockDim.x + threadIdx.x;  // over E*H
  if (i >= (long)E * H) return;
  int eid = (int)(i / H), hd = (int)(i % H);
  int s = src[eid], d = dst[eid];
  float e = as[s * H + hd] + ad[d * H + hd];
  e = (e > 0.f) ? e : NEG_SLOPE * e;
  ebuf[i] = e;
  atomicMax(&mflip[d * H + hd], flip_f32(e));
}

// m fixup in place: sentinel 0 (no incoming edges) -> 0.0, matching
// m = where(isfinite(m), m, 0) in the reference.
__global__ void segmax_fix(unsigned* __restrict__ m, long n) {
  long i = (long)blockIdx.x * blockDim.x + threadIdx.x;
  if (i >= n) return;
  unsigned u = m[i];
  ((float*)m)[i] = (u == 0u) ? 0.f : unflip_f32(u);
}

// -------- segment sum of exp(e - m[dst]) -----------------------------------
__global__ void edge_partition(const int* __restrict__ dst,
                               const float* __restrict__ ebuf,
                               const float* __restrict__ m,
                               float* __restrict__ z, int E, int H) {
  long i = (long)blockIdx.x * blockDim.x + threadIdx.x;  // over E*H
  if (i >= (long)E * H) return;
  int eid = (int)(i / H), hd = (int)(i % H);
  int d = dst[eid];
  atomicAdd(&z[d * H + hd], __expf(ebuf[i] - m[d * H + hd]));
}

// -------- weighted scatter-add of messages ---------------------------------
__global__ void edge_aggregate(const int* __restrict__ src,
                               const int* __restrict__ dst,
                               const float* __restrict__ ebuf,
                               const float* __restrict__ m,
                               const float* __restrict__ z,
                               const float* __restrict__ h,
                               float* __restrict__ agg,
                               int E, int H, int C, int CV) {
  long i = (long)blockIdx.x * blockDim.x + threadIdx.x;
  int nchunk = C / CV;
  long total = (long)E * H * nchunk;
  if (i >= total) return;
  int cc = (int)(i % nchunk);
  long t = i / nchunk;
  int hd  = (int)(t % H);
  int eid = (int)(t / H);
  int s = src[eid], d = dst[eid];
  float ev = ebuf[(long)eid * H + hd];
  float alpha = __expf(ev - m[d * H + hd]) / (z[d * H + hd] + EPS_F);
  const float* hp = h + ((size_t)s * H + hd) * C + cc * CV;
  float* op = agg + ((size_t)d * H + hd) * C + cc * CV;
  for (int c = 0; c < CV; ++c) atomicAdd(&op[c], alpha * hp[c]);
}

// -------- bias (+ optional ReLU) -------------------------------------------
__global__ void bias_act(float* __restrict__ x, const float* __restrict__ b,
                         long n, int F, int relu) {
  long i = (long)blockIdx.x * blockDim.x + threadIdx.x;  // over n*F
  if (i >= n * (long)F) return;
  float v = x[i] + b[i % F];
  x[i] = (relu && v < 0.f) ? 0.f : v;
}

// ---------------------------------------------------------------------------
struct LayerBufs {
  float *hbuf, *ebuf, *as, *ad, *m, *z;
};

static void run_layer(const float* in, int K, const float* Bm, int NCpad,
                      const float* a_s, const float* a_d, const float* bias,
                      int H, int C, const int* src, const int* dst,
                      float* aggOut, const LayerBufs& B, int relu,
                      hipStream_t stream) {
  const int F = H * C;               // NCout
  const long NH = (long)N_NODES * H;
  const long EH = (long)N_EDGES * H;

  // 1) h = in @ W  (WMMA; B panel staged through LDS)
  dim3 grid((unsigned)cdivl(N_NODES / 16, 8), (unsigned)(NCpad / 16));
  gemm_wmma_f16<<<grid, 256, 0, stream>>>(in, Bm, B.hbuf, N_NODES, K, NCpad, F);

  // 2) attention coefficients
  attn_coef<<<(unsigned)cdivl(NH, 256), 256, 0, stream>>>(
      B.hbuf, a_s, a_d, B.as, B.ad, N_NODES, H, C);

  // 3) init m (flip-encoded max), z, aggregation target
  fill_u32<<<(unsigned)cdivl(NH, 256), 256, 0, stream>>>((unsigned*)B.m, 0u, NH);
  fill_u32<<<(unsigned)cdivl(NH, 256), 256, 0, stream>>>((unsigned*)B.z, 0u, NH);
  fill_u32<<<(unsigned)cdivl((long)N_NODES * F, 256), 256, 0, stream>>>(
      (unsigned*)aggOut, 0u, (long)N_NODES * F);

  // 4) edge logits + segment max
  edge_logits<<<(unsigned)cdivl(EH, 256), 256, 0, stream>>>(
      src, dst, B.as, B.ad, B.ebuf, (unsigned*)B.m, N_EDGES, H);
  segmax_fix<<<(unsigned)cdivl(NH, 256), 256, 0, stream>>>((unsigned*)B.m, NH);

  // 5) softmax partition
  edge_partition<<<(unsigned)cdivl(EH, 256), 256, 0, stream>>>(
      dst, B.ebuf, B.m, B.z, N_EDGES, H);

  // 6) weighted aggregation
  int CV = (C % 4 == 0) ? 4 : C;
  long tot = EH * (C / CV);
  edge_aggregate<<<(unsigned)cdivl(tot, 256), 256, 0, stream>>>(
      src, dst, B.ebuf, B.m, B.z, B.hbuf, aggOut, N_EDGES, H, C, CV);

  // 7) bias (+ReLU)
  bias_act<<<(unsigned)cdivl((long)N_NODES * F, 256), 256, 0, stream>>>(
      aggOut, bias, N_NODES, F, relu);
}

extern "C" void kernel_launch(void* const* d_in, const int* in_sizes, int n_in,
                              void* d_out, int out_size, void* d_ws, size_t ws_size,
                              hipStream_t stream) {
  const float* x   = (const float*)d_in[0];
  const int*   ei  = (const int*)d_in[1];
  const int*   src = ei;
  const int*   dst = ei + N_EDGES;
  const float* W1 = (const float*)d_in[2],  *as1 = (const float*)d_in[3],
             *ad1 = (const float*)d_in[4],  *b1  = (const float*)d_in[5];
  const float* W2 = (const float*)d_in[6],  *as2 = (const float*)d_in[7],
             *ad2 = (const float*)d_in[8],  *b2  = (const float*)d_in[9];
  const float* W3 = (const float*)d_in[10], *as3 = (const float*)d_in[11],
             *ad3 = (const float*)d_in[12], *b3  = (const float*)d_in[13];

  // workspace carve-up (256B aligned slices)
  char* ws = (char*)d_ws;
  size_t off = 0;
  auto carve = [&](size_t bytes) -> void* {
    void* p = ws + off;
    off = (off + bytes + 255) & ~((size_t)255);
    return p;
  };
  float* hbuf  = (float*)carve((size_t)N_NODES * 256 * 4);  // GEMM output h
  float* agg   = (float*)carve((size_t)N_NODES * 256 * 4);  // aggregated output
  float* ebuf  = (float*)carve((size_t)N_EDGES * 4 * 4);    // edge logits [E,H]
  float* as    = (float*)carve((size_t)N_NODES * 4 * 4);
  float* ad    = (float*)carve((size_t)N_NODES * 4 * 4);
  float* m     = (float*)carve((size_t)N_NODES * 4 * 4);
  float* z     = (float*)carve((size_t)N_NODES * 4 * 4);
  float* w3pad = (float*)carve((size_t)256 * 16 * 4);       // W3 zero-padded
  (void)ws_size; (void)in_sizes; (void)n_in; (void)out_size;

  LayerBufs B{hbuf, ebuf, as, ad, m, z};

  // Pad W3 [256,2] -> [256,16] once per call (deterministic, graph-safe).
  pad_w<<<(unsigned)cdivl(256 * 16, 256), 256, 0, stream>>>(W3, w3pad, 256, 2, 16);

  // Layer 1: 128 -> 64 x 4 heads, ReLU
  run_layer(x, 128, W1, 256, as1, ad1, b1, 4, 64, src, dst, agg, B, 1, stream);
  // Layer 2: 256 -> 64 x 4 heads, ReLU  (reads agg, aggregates back into agg)
  run_layer(agg, 256, W2, 256, as2, ad2, b2, 4, 64, src, dst, agg, B, 1, stream);
  // Layer 3: 256 -> 2 x 1 head, no activation, aggregate straight into d_out
  run_layer(agg, 256, w3pad, 16, as3, ad3, b3, 1, 2, src, dst, (float*)d_out, B, 0, stream);
}
